// StoichNet_15616501088590
// MI455X (gfx1250) — compile-verified, matrix operand order checked by reading, not run on hardware
//
#include <hip/hip_runtime.h>

// ---------------------------------------------------------------------------
// StoichNet fused forward for MI455X (gfx1250, wave32, WMMA bf16)
// ---------------------------------------------------------------------------

typedef __attribute__((ext_vector_type(16))) __bf16 v16bf;
typedef __attribute__((ext_vector_type(8)))  __bf16 v8bf;
typedef __attribute__((ext_vector_type(8)))  float  v8f;

#define NELEM  262144
#define CRXN   65536
#define EFEA   200
#define RFEA   128
#define NHEAD  4

#define TM      64      // rows per block tile
#define THREADS 128     // 4 waves (wave32)
#define NWAVES  4
#define MT      4       // 16-row m-tiles per block (TM/16)

#define S0 360          // LDS stride (halves) for wide buffer  (352 padded, conflict-free)
#define S1 264          // LDS stride (halves) for narrow buffer (256 padded, conflict-free)
#define LDS_BYTES (TM*S0*2 + TM*S1*2 + TM*4)

// packed bf16 weight offsets in halves: [tensor][head][ntile][ktile][lane][16]
#define OW0   0LL
#define OW1   360448LL
#define OW2   622592LL
#define OW3   884736LL
#define OW4   1146880LL
#define OW5   1277952LL
#define OW6   1343488LL
#define ORW0  1376256LL
#define ORW4  1736704LL
#define ORW6  1867776LL
#define PACK_HALVES 1900544LL
#define PH_W0  90112LL
#define PH_W1  65536LL
#define PH_W4  32768LL
#define PH_W5  16384LL
#define PH_W6  8192LL
#define PH_RW0 90112LL
#define PH_RW4 32768LL
#define PH_RW6 8192LL

#define PACK_BYTES (PACK_HALVES * 2)           // 3,801,088
#define GATE_BYTES ((long long)NHEAD * NELEM * 4)  // 4,194,304

static __device__ __forceinline__ unsigned short f2bf(float f) {
  unsigned int u = __float_as_uint(f);
  u += 0x7FFFu + ((u >> 16) & 1u);             // round-to-nearest-even
  return (unsigned short)(u >> 16);
}
static __device__ __forceinline__ float bf2f(unsigned short h) {
  return __uint_as_float(((unsigned int)h) << 16);
}
static __device__ __forceinline__ void atomicMaxF(float* addr, float v) {
  if (v >= 0.0f) atomicMax((int*)addr, __float_as_int(v));
  else           atomicMin((unsigned int*)addr, __float_as_uint(v));
}

// ---------------------------------------------------------------------------
// init segment-softmax scratch: segMax = -inf, segSum = 0
// ---------------------------------------------------------------------------
__global__ void init_seg_kernel(float* __restrict__ segMax, float* __restrict__ segSum) {
  int t = blockIdx.x * blockDim.x + threadIdx.x;
  if (t < CRXN * NHEAD) {
    segMax[t] = __uint_as_float(0xFF800000u);
    segSum[t] = 0.0f;
  }
}

// ---------------------------------------------------------------------------
// pack f32 weights (H, dout, din) -> bf16 WMMA B-fragment layout
// lane<16 : col n = lane,    k = kbase + {0..7, 16..23}
// lane>=16: col n = lane-16, k = kbase + {8..15, 24..31}
// ---------------------------------------------------------------------------
__global__ void pack_w_kernel(const float* __restrict__ src, unsigned short* __restrict__ dst,
                              int dout, int din, int ktiles) {
  int ntiles = dout >> 4;
  long long perHead = (long long)ntiles * ktiles * 512;
  long long total = perHead * NHEAD;
  long long p = (long long)blockIdx.x * blockDim.x + threadIdx.x;
  if (p >= total) return;
  int j    = (int)(p & 15);
  int lane = (int)((p >> 4) & 31);
  long long q = p >> 9;
  int kt = (int)(q % ktiles); q /= ktiles;
  int nt = (int)(q % ntiles); q /= ntiles;
  int h  = (int)q;
  int n  = nt * 16 + (lane & 15);
  int kbase = kt * 32 + ((lane & 16) ? 8 : 0);
  int k = kbase + ((j < 8) ? j : j + 8);
  float v = (k < din) ? src[((long long)h * dout + n) * din + k] : 0.0f;
  dst[p] = f2bf(v);
}

// ---------------------------------------------------------------------------
// A-fragment (16x32 bf16) from row-major bf16 LDS: two ds_load_b128 per lane
// ---------------------------------------------------------------------------
static __device__ __forceinline__ v16bf load_afrag(const unsigned short* lds, int stride,
                                                   int mtile, int kt, int lane) {
  int row = mtile * 16 + (lane & 15);
  int col = kt * 32 + ((lane & 16) ? 8 : 0);
  const unsigned short* p = lds + row * stride + col;
  v8bf lo = *(const v8bf*)p;        // K = kbase+0..7
  v8bf hi = *(const v8bf*)(p + 16); // K = kbase+16..23
  return __builtin_shufflevector(lo, hi, 0,1,2,3,4,5,6,7,8,9,10,11,12,13,14,15);
}

// ---------------------------------------------------------------------------
// one residual-MLP layer: out = relu(in @ W^T + b) + res
// res = in @ RW^T (rwPack != null) or identity (rwPack == null)
// ---------------------------------------------------------------------------
static __device__ __forceinline__ void mlp_layer(const unsigned short* inL, int inStride,
                                                 unsigned short* outL, int outStride,
                                                 int dinPad, int dout,
                                                 const unsigned short* wPack,
                                                 const unsigned short* rwPack,
                                                 const float* bias,
                                                 int lane, int wave) {
  const int ktiles = dinPad >> 5;
  const int ntiles = dout >> 4;
  const v8f vzero = {0.f,0.f,0.f,0.f,0.f,0.f,0.f,0.f};
  for (int nt = wave; nt < ntiles; nt += NWAVES) {
    v8f acc[MT], racc[MT];
    #pragma unroll
    for (int m = 0; m < MT; ++m) { acc[m] = vzero; racc[m] = vzero; }
    const unsigned short* wp  = wPack + (long long)nt * ktiles * 512;
    const unsigned short* rwp = rwPack ? rwPack + (long long)nt * ktiles * 512 : nullptr;
    for (int kt = 0; kt < ktiles; ++kt) {
      if (kt + 1 < ktiles) __builtin_prefetch(wp + (kt + 1) * 512 + lane * 16, 0, 3);
      v16bf bfrag = *(const v16bf*)(wp + kt * 512 + lane * 16);
      v16bf rfrag;
      if (rwp) rfrag = *(const v16bf*)(rwp + kt * 512 + lane * 16);
      #pragma unroll
      for (int m = 0; m < MT; ++m) {
        v16bf afrag = load_afrag(inL, inStride, m, kt, lane);
        acc[m] = __builtin_amdgcn_wmma_f32_16x16x32_bf16(
            false, afrag, false, bfrag, (short)0, acc[m], false, false);
        if (rwp)
          racc[m] = __builtin_amdgcn_wmma_f32_16x16x32_bf16(
              false, afrag, false, rfrag, (short)0, racc[m], false, false);
      }
    }
    // epilogue: C/D layout -> lanes 0-15: M=r, lanes 16-31: M=8+r, N=lane%16
    const int n = nt * 16 + (lane & 15);
    const float bn = bias[n];
    #pragma unroll
    for (int m = 0; m < MT; ++m) {
      const int mbase = m * 16 + ((lane & 16) ? 8 : 0);
      #pragma unroll
      for (int r = 0; r < 8; ++r) {
        const int row = mbase + r;
        float z = acc[m][r] + bn;
        float v = z > 0.f ? z : 0.f;
        if (rwp) v += racc[m][r];
        else     v += bf2f(inL[row * inStride + n]);  // identity residual (dout==din)
        outL[row * outStride + n] = f2bf(v);
      }
    }
  }
}

// ---------------------------------------------------------------------------
// fused: gather+concat -> 7 residual layers (WMMA) -> gate -> atomic seg-max
// ---------------------------------------------------------------------------
__global__ void __launch_bounds__(THREADS)
fused_mlp_kernel(const float* __restrict__ elemFea,
                 const int*   __restrict__ rxnIdx,
                 const float* __restrict__ rxnEmb,
                 const float* __restrict__ b0, const float* __restrict__ b1,
                 const float* __restrict__ b2, const float* __restrict__ b3,
                 const float* __restrict__ b4, const float* __restrict__ b5,
                 const float* __restrict__ b6,
                 const float* __restrict__ wout, const float* __restrict__ bout,
                 const unsigned short* __restrict__ packW,
                 float* __restrict__ gateBuf, float* __restrict__ segMax) {
  extern __shared__ unsigned char smem[];
  unsigned short* buf0 = (unsigned short*)smem;                       // stride S0
  unsigned short* buf1 = (unsigned short*)(smem + TM * S0 * 2);       // stride S1
  int* idxL = (int*)(smem + TM * S0 * 2 + TM * S1 * 2);

  const int tid  = threadIdx.x;
  const int lane = tid & 31;
  const int wave = tid >> 5;
  const int head = blockIdx.y;
  const long long rowBase = (long long)blockIdx.x * TM;

  if (tid < TM) idxL[tid] = rxnIdx[rowBase + tid];
  __syncthreads();

  // phase 0: gather + concat + zero-pad into buf0 (bf16, row-major, stride S0)
  for (int p = tid; p < TM * S0; p += THREADS) {
    int r = p / S0, c = p - r * S0;
    float v;
    if (c < EFEA)             v = elemFea[(rowBase + r) * EFEA + c];
    else if (c < EFEA + RFEA) v = rxnEmb[(long long)idxL[r] * RFEA + (c - EFEA)];
    else                      v = 0.0f;  // K-pad 328..351 (and stride pad)
    buf0[r * S0 + c] = f2bf(v);
  }
  __syncthreads();

  const unsigned short* P = packW;
  // L0: 352 -> 256, projection residual
  mlp_layer(buf0, S0, buf1, S1, 352, 256, P + OW0 + head * PH_W0,
            P + ORW0 + head * PH_RW0, b0 + head * 256, lane, wave);
  __syncthreads();
  // L1..L3: 256 -> 256, identity residual
  mlp_layer(buf1, S1, buf0, S1, 256, 256, P + OW1 + head * PH_W1, nullptr,
            b1 + head * 256, lane, wave);
  __syncthreads();
  mlp_layer(buf0, S1, buf1, S1, 256, 256, P + OW2 + head * PH_W1, nullptr,
            b2 + head * 256, lane, wave);
  __syncthreads();
  mlp_layer(buf1, S1, buf0, S1, 256, 256, P + OW3 + head * PH_W1, nullptr,
            b3 + head * 256, lane, wave);
  __syncthreads();
  // L4: 256 -> 128, projection residual
  mlp_layer(buf0, S1, buf1, S1, 256, 128, P + OW4 + head * PH_W4,
            P + ORW4 + head * PH_RW4, b4 + head * 128, lane, wave);
  __syncthreads();
  // L5: 128 -> 128, identity residual
  mlp_layer(buf1, S1, buf0, S1, 128, 128, P + OW5 + head * PH_W5, nullptr,
            b5 + head * 128, lane, wave);
  __syncthreads();
  // L6: 128 -> 64, projection residual
  mlp_layer(buf0, S1, buf1, S1, 128, 64, P + OW6 + head * PH_W6,
            P + ORW6 + head * PH_RW6, b6 + head * 64, lane, wave);
  __syncthreads();

  // gate = h @ wout^T + bout, per row; then atomic segment max
  if (tid < TM) {
    const float* wo = wout + head * 64;
    float g = bout[head];
    #pragma unroll 8
    for (int k = 0; k < 64; ++k) g += bf2f(buf1[tid * S1 + k]) * wo[k];
    long long gi = rowBase + tid;
    gateBuf[(long long)head * NELEM + gi] = g;
    atomicMaxF(&segMax[(long long)idxL[tid] * NHEAD + head], g);
  }
}

// ---------------------------------------------------------------------------
// segment softmax pass 2: e = exp(g - max); atomicAdd into segment sums
// ---------------------------------------------------------------------------
__global__ void exp_sum_kernel(const int* __restrict__ rxnIdx,
                               const float* __restrict__ segMax,
                               float* __restrict__ gateBuf,
                               float* __restrict__ segSum) {
  long long t = (long long)blockIdx.x * blockDim.x + threadIdx.x;
  if (t >= (long long)NHEAD * NELEM) return;
  int h = (int)(t / NELEM);
  long long n = t - (long long)h * NELEM;
  int c = rxnIdx[n];
  float e = expf(gateBuf[t] - segMax[(long long)c * NHEAD + h]);
  gateBuf[t] = e;
  atomicAdd(&segSum[(long long)c * NHEAD + h], e);
}

// ---------------------------------------------------------------------------
// segment softmax pass 3: normalize + mean over heads
// ---------------------------------------------------------------------------
__global__ void finalize_kernel(const int* __restrict__ rxnIdx,
                                const float* __restrict__ gateBuf,
                                const float* __restrict__ segSum,
                                float* __restrict__ out) {
  long long n = (long long)blockIdx.x * blockDim.x + threadIdx.x;
  if (n >= NELEM) return;
  int c = rxnIdx[n];
  float s = 0.f;
  #pragma unroll
  for (int h = 0; h < NHEAD; ++h)
    s += gateBuf[(long long)h * NELEM + n] / (segSum[(long long)c * NHEAD + h] + 1e-13f);
  out[n] = 0.25f * s;
}

// ---------------------------------------------------------------------------
extern "C" void kernel_launch(void* const* d_in, const int* in_sizes, int n_in,
                              void* d_out, int out_size, void* d_ws, size_t ws_size,
                              hipStream_t stream) {
  (void)in_sizes; (void)n_in; (void)out_size; (void)ws_size;
  const float* elemFea = (const float*)d_in[0];
  const int*   rxnIdx  = (const int*)d_in[1];
  const float* rxnEmb  = (const float*)d_in[2];
  const float* w[7];  const float* b[7];
  for (int i = 0; i < 7; ++i) { w[i] = (const float*)d_in[3 + 2*i]; b[i] = (const float*)d_in[4 + 2*i]; }
  const float* rw0  = (const float*)d_in[17];
  const float* rw4  = (const float*)d_in[18];
  const float* rw6  = (const float*)d_in[19];
  const float* wout = (const float*)d_in[20];
  const float* bout = (const float*)d_in[21];
  float* out = (float*)d_out;

  char* base = (char*)d_ws;
  unsigned short* packW = (unsigned short*)base;
  float* gateBuf = (float*)(base + PACK_BYTES);
  float* segMax  = (float*)(base + PACK_BYTES + GATE_BYTES);
  float* segSum  = segMax + (long long)CRXN * NHEAD;

  // init segment scratch
  init_seg_kernel<<<(CRXN * NHEAD + 255) / 256, 256, 0, stream>>>(segMax, segSum);

  // pack all weights to bf16 B-fragment layout (each tiny vs GEMM work)
  struct PackJob { const float* src; long long off; int dout, din, kt; };
  const PackJob jobs[10] = {
    { w[0], OW0, 256, 328, 11 }, { w[1], OW1, 256, 256, 8 },
    { w[2], OW2, 256, 256, 8 },  { w[3], OW3, 256, 256, 8 },
    { w[4], OW4, 128, 256, 8 },  { w[5], OW5, 128, 128, 4 },
    { w[6], OW6,  64, 128, 4 },  { rw0, ORW0, 256, 328, 11 },
    { rw4, ORW4, 128, 256, 8 },  { rw6, ORW6,  64, 128, 4 },
  };
  for (int i = 0; i < 10; ++i) {
    long long total = (long long)NHEAD * (jobs[i].dout >> 4) * jobs[i].kt * 512;
    pack_w_kernel<<<(int)((total + 255) / 256), 256, 0, stream>>>(
        jobs[i].src, packW + jobs[i].off, jobs[i].dout, jobs[i].din, jobs[i].kt);
  }

  // fused WMMA MLP + gate + segment max
  (void)hipFuncSetAttribute((const void*)fused_mlp_kernel,
                            hipFuncAttributeMaxDynamicSharedMemorySize, LDS_BYTES);
  fused_mlp_kernel<<<dim3(NELEM / TM, NHEAD), THREADS, LDS_BYTES, stream>>>(
      elemFea, rxnIdx, rxnEmb, b[0], b[1], b[2], b[3], b[4], b[5], b[6],
      wout, bout, packW, gateBuf, segMax);

  // segment softmax
  exp_sum_kernel<<<(NHEAD * NELEM + 255) / 256, 256, 0, stream>>>(rxnIdx, segMax, gateBuf, segSum);
  finalize_kernel<<<(NELEM + 255) / 256, 256, 0, stream>>>(rxnIdx, gateBuf, segSum, out);
}
